// TransformerBlock_39032662786707
// MI455X (gfx1250) — matrix operand order repaired and checked by reading, not compiled, not run on hardware
//
#include <hip/hip_runtime.h>
#include <hip/hip_bf16.h>
#include <math.h>

// Transformer block for gfx1250 (MI455X) using bf16 WMMA (v_wmma_f32_16x16x32_bf16).
// Shapes: L=2048, B=2, D=1024, H=16, HD=64, DFF=4096.

#define L_SEQ  2048
#define BATCH  2
#define DMODEL 1024
#define NHEAD  16
#define HDIM   64
#define DFFN   4096
#define NTOK   (L_SEQ * BATCH)   // 4096 token rows, row index = l*B + b

typedef __attribute__((ext_vector_type(16))) __bf16 v16bf;
typedef __attribute__((ext_vector_type(8)))  float  v8f;

union V16 { v16bf v; uint4 q[2]; };

__device__ __forceinline__ unsigned short f2bf(float f) {
  unsigned u = __float_as_uint(f);
  u += 0x7FFFu + ((u >> 16) & 1u);        // round-to-nearest-even
  return (unsigned short)(u >> 16);
}

// reuse_a hint must be an integer-constant expression -> template param.
template <bool RA>
__device__ __forceinline__ v8f wmma_bf16(v16bf a, v16bf b, v8f c) {
  return __builtin_amdgcn_wmma_f32_16x16x32_bf16(false, a, false, b,
                                                 (short)0, c, RA, false);
}

// ---------------------------------------------------------------- layernorm
// one block per row of D=1024, 256 threads * 4 elems, bf16 output
__global__ void __launch_bounds__(256)
k_layernorm_bf16(const float* __restrict__ x, const float* __restrict__ g,
                 const float* __restrict__ bta, unsigned short* __restrict__ out) {
  __shared__ float s_sum[256], s_sq[256];
  int row = blockIdx.x;
  const float* xr = x + (size_t)row * DMODEL;
  float4 v = *(const float4*)(xr + threadIdx.x * 4);
  s_sum[threadIdx.x] = v.x + v.y + v.z + v.w;
  s_sq[threadIdx.x]  = v.x*v.x + v.y*v.y + v.z*v.z + v.w*v.w;
  __syncthreads();
  for (int off = 128; off > 0; off >>= 1) {
    if ((int)threadIdx.x < off) {
      s_sum[threadIdx.x] += s_sum[threadIdx.x + off];
      s_sq[threadIdx.x]  += s_sq[threadIdx.x + off];
    }
    __syncthreads();
  }
  float mu  = s_sum[0] * (1.0f / DMODEL);
  float var = s_sq[0] * (1.0f / DMODEL) - mu * mu;
  float rs  = rsqrtf(var + 1e-5f);
  float e[4] = {v.x, v.y, v.z, v.w};
  unsigned short* orow = out + (size_t)row * DMODEL;
  #pragma unroll
  for (int j = 0; j < 4; ++j) {
    int c = threadIdx.x * 4 + j;
    orow[c] = f2bf((e[j] - mu) * rs * g[c] + bta[c]);
  }
}

// ------------------------------------------------- weight cast + transpose
// src: f32 [K, N] row-major  ->  dst: bf16 [N, K] row-major
__global__ void __launch_bounds__(256)
k_transpose_bf16(const float* __restrict__ w, unsigned short* __restrict__ wt,
                 int K, int N) {
  size_t id = (size_t)blockIdx.x * 256 + threadIdx.x;
  int n = (int)(id / K);
  int k = (int)(id % K);
  wt[(size_t)n * K + k] = f2bf(w[(size_t)k * N + n]);
}

// ----------------------------------------------------------- WMMA GEMM
// C[M,N] = epilogue( A_bf16[M,K] @ Bt_bf16[N,K]^T + bias[N] (+ residual) )
// register-blocked: one wave = 64x64 output (4x4 WMMA tiles), 8 waves/block
// cover 128x256.  Per 32-k-step: 16 b128 loads feed 16 WMMAs (32 flop/byte).
__global__ void __launch_bounds__(256)
k_gemm_bf16(const unsigned short* __restrict__ A,
            const unsigned short* __restrict__ Bt,
            const float* __restrict__ bias,
            const float* __restrict__ residual,   // [M,N] or null
            float* __restrict__ outF,             // [M,N] or null
            unsigned short* __restrict__ outB,    // [M,N] or null
            int M, int N, int K, int gelu) {
  int lane = threadIdx.x & 31;
  int wave = threadIdx.x >> 5;
  int half = lane >> 4;
  int l15  = lane & 15;

  int bTilesN = N >> 8;                        // blocks along N (256 wide)
  int bm = blockIdx.x / bTilesN;
  int bn = blockIdx.x % bTilesN;
  int wm = wave >> 2, wn = wave & 3;           // 2x4 waves in block
  int rowBase = bm * 128 + wm * 64;
  int colBase = bn * 256 + wn * 64;

  const unsigned short* Ar[4];
  const unsigned short* Br[4];
  #pragma unroll
  for (int i = 0; i < 4; ++i) {
    Ar[i] = A  + (size_t)(rowBase + i * 16 + l15) * K;  // A frag: row = lane&15
    Br[i] = Bt + (size_t)(colBase + i * 16 + l15) * K;  // B frag: col = lane&15
  }

  v8f acc[4][4] = {};
  for (int kk = 0; kk < K; kk += 32) {
    V16 a[4], b[4];
    #pragma unroll
    for (int i = 0; i < 4; ++i) {
      // A 16x32 bf16 layout: elem i -> K = half*16 + i
      a[i].q[0] = *(const uint4*)(Ar[i] + kk + half * 16);
      a[i].q[1] = *(const uint4*)(Ar[i] + kk + half * 16 + 8);
      // B 32x16 bf16 layout: elem i -> K = half*16 + i  (contiguous in [N,K])
      b[i].q[0] = *(const uint4*)(Br[i] + kk + half * 16);
      b[i].q[1] = *(const uint4*)(Br[i] + kk + half * 16 + 8);
    }
    // speculative prefetch 2 k-steps ahead (global_prefetch_b8; OOB is dropped)
    #pragma unroll
    for (int i = 0; i < 4; ++i) {
      __builtin_prefetch((const void*)(Ar[i] + kk + 64), 0, 3);
      __builtin_prefetch((const void*)(Br[i] + kk + 64), 0, 3);
    }
    #pragma unroll
    for (int i = 0; i < 4; ++i) {
      // reuse_a: the next 3 WMMAs in the burst reuse the same A fragment
      acc[i][0] = wmma_bf16<true >(a[i].v, b[0].v, acc[i][0]);
      acc[i][1] = wmma_bf16<true >(a[i].v, b[1].v, acc[i][1]);
      acc[i][2] = wmma_bf16<true >(a[i].v, b[2].v, acc[i][2]);
      acc[i][3] = wmma_bf16<false>(a[i].v, b[3].v, acc[i][3]);
    }
  }

  #pragma unroll
  for (int j = 0; j < 4; ++j) {
    int col = colBase + j * 16 + l15;
    float bv = bias[col];
    #pragma unroll
    for (int i = 0; i < 4; ++i) {
      #pragma unroll
      for (int ii = 0; ii < 8; ++ii) {
        int row = rowBase + i * 16 + ii + half * 8;   // C layout: M = ii + 8*half
        float v = acc[i][j][ii] + bv;
        if (residual) v += residual[(size_t)row * N + col];
        if (gelu) {  // tanh-approx GELU (jax.nn.gelu default)
          float u = 0.7978845608028654f * (v + 0.044715f * v * v * v);
          v = 0.5f * v * (1.0f + tanhf(u));
        }
        if (outF) outF[(size_t)row * N + col] = v;
        if (outB) outB[(size_t)row * N + col] = f2bf(v);
      }
    }
  }
}

// ------------------------------------------------------------- QKV split
// qkv f32 [NTOK, 3D] -> q bf16 (scaled 1/8) / k bf16 in [B,H,L,64],
// vT bf16 in [B,H,64,L], and f32 k/v into `present` region of d_out.
__global__ void __launch_bounds__(256)
k_split_qkv(const float* __restrict__ qkv,
            unsigned short* __restrict__ qb, unsigned short* __restrict__ kb,
            unsigned short* __restrict__ vtb,
            float* __restrict__ kout, float* __restrict__ vout) {
  int id  = blockIdx.x * 256 + threadIdx.x;   // NTOK*DMODEL threads
  int row = id >> 10;                          // token row = l*B + b
  int c   = id & (DMODEL - 1);
  int l = row / BATCH, b = row % BATCH;
  int h = c >> 6, hd = c & 63;
  const float* base = qkv + (size_t)row * (3 * DMODEL);
  float qv = base[c];
  float kv = base[DMODEL + c];
  float vv = base[2 * DMODEL + c];
  size_t off = (((size_t)(b * NHEAD + h)) * L_SEQ + l) * HDIM + hd;
  qb[off] = f2bf(qv * 0.125f);                 // fold 1/sqrt(64) exactly
  kb[off] = f2bf(kv);
  kout[off] = kv;
  vout[off] = vv;
  vtb[(((size_t)(b * NHEAD + h)) * HDIM + hd) * L_SEQ + l] = f2bf(vv);
}

// ------------------------------------------------------ flash attention
// one wave = one (b,h, 16-query tile); online softmax over 32-key steps.
__global__ void __launch_bounds__(128)
k_attn(const unsigned short* __restrict__ qb,
       const unsigned short* __restrict__ kb,
       const unsigned short* __restrict__ vtb,
       unsigned short* __restrict__ attn_out) {
  __shared__ unsigned short plds[4][16 * 32];
  int lane = threadIdx.x & 31, wave = threadIdx.x >> 5;
  int half = lane >> 4, l15 = lane & 15;
  const int QT = L_SEQ / 16;                   // 128 q tiles per head
  int id = blockIdx.x * 4 + wave;
  int bh = id / QT, qt = id % QT;
  const unsigned short* Qh = qb  + (size_t)bh * L_SEQ * HDIM;
  const unsigned short* Kh = kb  + (size_t)bh * L_SEQ * HDIM;
  const unsigned short* Vt = vtb + (size_t)bh * HDIM * L_SEQ;

  V16 aq0, aq1;                                // Q tile, contraction chunks 0-31 / 32-63
  {
    const unsigned short* qr = Qh + (size_t)(qt * 16 + l15) * HDIM;
    aq0.q[0] = *(const uint4*)(qr + half * 16);
    aq0.q[1] = *(const uint4*)(qr + half * 16 + 8);
    aq1.q[0] = *(const uint4*)(qr + 32 + half * 16);
    aq1.q[1] = *(const uint4*)(qr + 32 + half * 16 + 8);
  }

  v8f o[4] = {};                               // 16 x 64 output in f32
  float mi[8], li[8];
  #pragma unroll
  for (int i = 0; i < 8; ++i) { mi[i] = -1e30f; li[i] = 0.0f; }

  unsigned short* pw = plds[wave];

  for (int t = 0; t < L_SEQ / 32; ++t) {
    v8f s0 = {}, s1 = {};
    {   // keys t*32 .. t*32+15
      const unsigned short* kr = Kh + (size_t)(t * 32 + l15) * HDIM;
      V16 b0, b1;
      b0.q[0] = *(const uint4*)(kr + half * 16);
      b0.q[1] = *(const uint4*)(kr + half * 16 + 8);
      b1.q[0] = *(const uint4*)(kr + 32 + half * 16);
      b1.q[1] = *(const uint4*)(kr + 32 + half * 16 + 8);
      s0 = wmma_bf16<false>(aq0.v, b0.v, s0);
      s0 = wmma_bf16<false>(aq1.v, b1.v, s0);
    }
    {   // keys t*32+16 .. t*32+31
      const unsigned short* kr = Kh + (size_t)(t * 32 + 16 + l15) * HDIM;
      V16 b0, b1;
      b0.q[0] = *(const uint4*)(kr + half * 16);
      b0.q[1] = *(const uint4*)(kr + half * 16 + 8);
      b1.q[0] = *(const uint4*)(kr + 32 + half * 16);
      b1.q[1] = *(const uint4*)(kr + 32 + half * 16 + 8);
      s1 = wmma_bf16<false>(aq0.v, b0.v, s1);
      s1 = wmma_bf16<false>(aq1.v, b1.v, s1);
    }

    // online softmax; each row of the 16x32 S lives on one 16-lane half-wave
    #pragma unroll
    for (int i = 0; i < 8; ++i) {
      float r = fmaxf(s0[i], s1[i]);
      r = fmaxf(r, __shfl_xor(r, 1, 16));
      r = fmaxf(r, __shfl_xor(r, 2, 16));
      r = fmaxf(r, __shfl_xor(r, 4, 16));
      r = fmaxf(r, __shfl_xor(r, 8, 16));
      float nm = fmaxf(mi[i], r);
      float p0 = __expf(s0[i] - nm);
      float p1 = __expf(s1[i] - nm);
      float corr = __expf(mi[i] - nm);
      mi[i] = nm;
      float rs = p0 + p1;
      rs += __shfl_xor(rs, 1, 16);
      rs += __shfl_xor(rs, 2, 16);
      rs += __shfl_xor(rs, 4, 16);
      rs += __shfl_xor(rs, 8, 16);
      li[i] = li[i] * corr + rs;
      o[0][i] *= corr; o[1][i] *= corr; o[2][i] *= corr; o[3][i] *= corr;
      int prow = i + half * 8;                 // C-layout row -> LDS row-major 16x32
      pw[prow * 32 + l15]      = f2bf(p0);
      pw[prow * 32 + 16 + l15] = f2bf(p1);
    }
    __asm volatile("" ::: "memory");           // same-wave LDS ops are HW in-order

    V16 ap;                                    // reload P in A-fragment layout
    ap.q[0] = *(const uint4*)(pw + l15 * 32 + half * 16);
    ap.q[1] = *(const uint4*)(pw + l15 * 32 + half * 16 + 8);

    // O += P @ V, hd cols n*16..+15; manual unroll for constant reuse_a hints
#define PV_STEP(n, RA)                                                         \
    {                                                                          \
      V16 bv;                                                                  \
      const unsigned short* vr =                                               \
          Vt + (size_t)((n) * 16 + l15) * L_SEQ + t * 32 + half * 16;          \
      bv.q[0] = *(const uint4*)(vr);                                           \
      bv.q[1] = *(const uint4*)(vr + 8);                                       \
      o[n] = wmma_bf16<RA>(ap.v, bv.v, o[n]);                                  \
    }
    PV_STEP(0, true)
    PV_STEP(1, true)
    PV_STEP(2, true)
    PV_STEP(3, false)
#undef PV_STEP
  }

  int b = bh / NHEAD, h = bh % NHEAD;
  #pragma unroll
  for (int i = 0; i < 8; ++i) {
    float inv = 1.0f / li[i];
    int l = qt * 16 + i + half * 8;
    size_t rowoff = ((size_t)l * BATCH + b) * DMODEL + h * HDIM + l15;
    attn_out[rowoff +  0] = f2bf(o[0][i] * inv);
    attn_out[rowoff + 16] = f2bf(o[1][i] * inv);
    attn_out[rowoff + 32] = f2bf(o[2][i] * inv);
    attn_out[rowoff + 48] = f2bf(o[3][i] * inv);
  }
}

// ---------------------------------------------------------------- launch
extern "C" void kernel_launch(void* const* d_in, const int* in_sizes, int n_in,
                              void* d_out, int out_size, void* d_ws, size_t ws_size,
                              hipStream_t stream) {
  const float* x      = (const float*)d_in[0];
  const float* ln1_g  = (const float*)d_in[1];
  const float* ln1_b  = (const float*)d_in[2];
  const float* w_attn = (const float*)d_in[3];
  const float* b_attn = (const float*)d_in[4];
  const float* w_proj = (const float*)d_in[5];
  const float* b_proj = (const float*)d_in[6];
  const float* ln2_g  = (const float*)d_in[7];
  const float* ln2_b  = (const float*)d_in[8];
  const float* w_fc   = (const float*)d_in[9];
  const float* b_fc   = (const float*)d_in[10];
  const float* w_out  = (const float*)d_in[11];
  const float* b_out  = (const float*)d_in[12];

  float* out_x = (float*)d_out;                               // [L,B,D]
  float* kpres = out_x + (size_t)L_SEQ * BATCH * DMODEL;      // [B,H,L,64]
  float* vpres = kpres + (size_t)BATCH * NHEAD * L_SEQ * HDIM;

  char* ws = (char*)d_ws;
  size_t off = 0;
  auto alloc = [&](size_t bytes) {
    size_t o = off; off += (bytes + 255) & ~(size_t)255; return o;
  };
  unsigned short* h1    = (unsigned short*)(ws + alloc((size_t)NTOK * DMODEL * 2));
  unsigned short* wTat  = (unsigned short*)(ws + alloc((size_t)3 * DMODEL * DMODEL * 2));
  unsigned short* wTpr  = (unsigned short*)(ws + alloc((size_t)DMODEL * DMODEL * 2));
  unsigned short* wTfc  = (unsigned short*)(ws + alloc((size_t)DFFN * DMODEL * 2));
  unsigned short* wTou  = (unsigned short*)(ws + alloc((size_t)DMODEL * DFFN * 2));
  float*          qkv   = (float*)         (ws + alloc((size_t)NTOK * 3 * DMODEL * 4));
  unsigned short* qb    = (unsigned short*)(ws + alloc((size_t)BATCH * NHEAD * L_SEQ * HDIM * 2));
  unsigned short* kbuf  = (unsigned short*)(ws + alloc((size_t)BATCH * NHEAD * L_SEQ * HDIM * 2));
  unsigned short* vtb   = (unsigned short*)(ws + alloc((size_t)BATCH * NHEAD * HDIM * L_SEQ * 2));
  unsigned short* attnb = (unsigned short*)(ws + alloc((size_t)NTOK * DMODEL * 2));
  float*          xmid  = (float*)         (ws + alloc((size_t)NTOK * DMODEL * 4));
  unsigned short* h2    = (unsigned short*)(ws + alloc((size_t)NTOK * DMODEL * 2));
  unsigned short* act   = (unsigned short*)(ws + alloc((size_t)NTOK * DFFN * 2));

  // weight casts/transposes: f32 [K,N] -> bf16 [N,K]
  k_transpose_bf16<<<(DMODEL * 3 * DMODEL) / 256, 256, 0, stream>>>(w_attn, wTat, DMODEL, 3 * DMODEL);
  k_transpose_bf16<<<(DMODEL * DMODEL) / 256, 256, 0, stream>>>(w_proj, wTpr, DMODEL, DMODEL);
  k_transpose_bf16<<<(DMODEL * DFFN) / 256, 256, 0, stream>>>(w_fc, wTfc, DMODEL, DFFN);
  k_transpose_bf16<<<(DFFN * DMODEL) / 256, 256, 0, stream>>>(w_out, wTou, DFFN, DMODEL);

  // ln1 -> h1 (bf16)
  k_layernorm_bf16<<<NTOK, 256, 0, stream>>>(x, ln1_g, ln1_b, h1);

  // qkv = h1 @ w_attn + b_attn   [4096 x 3072]
  k_gemm_bf16<<<(NTOK / 128) * (3 * DMODEL / 256), 256, 0, stream>>>(
      h1, wTat, b_attn, nullptr, qkv, nullptr, NTOK, 3 * DMODEL, DMODEL, 0);

  // split into per-head q/k/vT (bf16) + present (f32)
  k_split_qkv<<<(NTOK * DMODEL) / 256, 256, 0, stream>>>(qkv, qb, kbuf, vtb, kpres, vpres);

  // flash attention -> attnb bf16 [NTOK, D]
  k_attn<<<(BATCH * NHEAD * (L_SEQ / 16)) / 4, 128, 0, stream>>>(qb, kbuf, vtb, attnb);

  // xmid = x + attnb @ w_proj + b_proj
  k_gemm_bf16<<<(NTOK / 128) * (DMODEL / 256), 256, 0, stream>>>(
      attnb, wTpr, b_proj, x, xmid, nullptr, NTOK, DMODEL, DMODEL, 0);

  // ln2 -> h2
  k_layernorm_bf16<<<NTOK, 256, 0, stream>>>(xmid, ln2_g, ln2_b, h2);

  // act = gelu(h2 @ w_fc + b_fc)  (bf16)
  k_gemm_bf16<<<(NTOK / 128) * (DFFN / 256), 256, 0, stream>>>(
      h2, wTfc, b_fc, nullptr, nullptr, act, NTOK, DFFN, DMODEL, 1);

  // out_x = xmid + act @ w_out + b_out
  k_gemm_bf16<<<(NTOK / 128) * (DMODEL / 256), 256, 0, stream>>>(
      act, wTou, b_out, xmid, out_x, nullptr, NTOK, DMODEL, DFFN, 0);
}
